// MoELayer_54735063220320
// MI455X (gfx1250) — compile-verified, matrix operand order checked by reading, not run on hardware
//
#include <hip/hip_runtime.h>
#include <hip/hip_bf16.h>

// ---------------------------------------------------------------------------
// MoE (8 experts, top-2, H=2048, N=8192) for gfx1250 (MI455X).
// Routed bf16 WMMA grouped-GEMM, double-buffered LDS staged with
// global_load_async_to_lds_b128 (ASYNCcnt) when available.
// ---------------------------------------------------------------------------

#define NUM_EXPERTS 8
#define HIDDEN      2048
#define N_TOKENS    8192
#define CAP         N_TOKENS     // worst-case tokens per expert

typedef __attribute__((ext_vector_type(16))) __bf16 v16bf;
typedef __attribute__((ext_vector_type(8)))  float  v8f;
typedef int v4i_vs __attribute__((vector_size(16)));   // matches builtin param type
typedef __attribute__((address_space(3))) v4i_vs* lds_v4i_ptr;

#if defined(__AMDGCN__) && __has_builtin(__builtin_amdgcn_global_load_async_to_lds_b128)
#define HAVE_ASYNC 1
#else
#define HAVE_ASYNC 0
#endif

__device__ __forceinline__ void wait_async_zero() {
#if HAVE_ASYNC
#if __has_builtin(__builtin_amdgcn_s_wait_asynccnt)
    __builtin_amdgcn_s_wait_asynccnt(0);
#else
    asm volatile("s_wait_asynccnt 0x0" ::: "memory");
#endif
#endif
}

__device__ __forceinline__ unsigned short f2bf(float f) {
    unsigned u = __float_as_uint(f);
    unsigned r = u + 0x7fffu + ((u >> 16) & 1u);   // round-to-nearest-even
    return (unsigned short)(r >> 16);
}

// ---------------------------------------------------------------- zero ------
__global__ void zero_f32(float* __restrict__ p, size_t n) {
    size_t i = (size_t)blockIdx.x * blockDim.x + threadIdx.x;
    size_t stride = (size_t)gridDim.x * blockDim.x;
    for (; i < n; i += stride) p[i] = 0.0f;
}

// ------------------------------------------------------- x fp32 -> bf16 -----
struct alignas(8) BF4 { unsigned short a, b, c, d; };

__global__ void convert_x(const float* __restrict__ x,
                          unsigned short* __restrict__ xb, size_t n4) {
    size_t i = (size_t)blockIdx.x * blockDim.x + threadIdx.x;
    size_t stride = (size_t)gridDim.x * blockDim.x;
    for (; i < n4; i += stride) {
        float4 v = ((const float4*)x)[i];
        BF4 o = { f2bf(v.x), f2bf(v.y), f2bf(v.z), f2bf(v.w) };
        ((BF4*)xb)[i] = o;
    }
}

// --------------------------------- We[e][k][n] -> Wt[e][n][k] (bf16) --------
__global__ void transpose_convert_w(const float* __restrict__ We,
                                    unsigned short* __restrict__ Wt) {
    __shared__ float t[32][33];
    int e  = blockIdx.z;
    int k0 = blockIdx.x * 32;
    int n0 = blockIdx.y * 32;
    const float* W = We + (size_t)e * HIDDEN * HIDDEN;
    for (int i = threadIdx.y; i < 32; i += 8)
        t[i][threadIdx.x] = W[(size_t)(k0 + i) * HIDDEN + n0 + threadIdx.x];
    __syncthreads();
    unsigned short* O = Wt + (size_t)e * HIDDEN * HIDDEN;
    for (int i = threadIdx.y; i < 32; i += 8)
        O[(size_t)(n0 + i) * HIDDEN + k0 + threadIdx.x] = f2bf(t[threadIdx.x][i]);
}

// ------------------------------------------------------------ gating --------
__global__ void gating_kernel(const float* __restrict__ x,
                              const float* __restrict__ Wg,
                              const float* __restrict__ bg,
                              int* __restrict__ counts,
                              int* __restrict__ tok_idx,
                              float* __restrict__ tok_w) {
    int gid  = blockIdx.x * blockDim.x + threadIdx.x;
    int tok  = gid >> 5;              // one wave32 per token
    int lane = threadIdx.x & 31;
    if (tok >= N_TOKENS) return;

    const float* xr = x + (size_t)tok * HIDDEN;
    float acc[NUM_EXPERTS];
#pragma unroll
    for (int e = 0; e < NUM_EXPERTS; ++e) acc[e] = 0.0f;

    for (int k = lane; k < HIDDEN; k += 32) {
        float xv = xr[k];
        const float* wr = Wg + (size_t)k * NUM_EXPERTS;
#pragma unroll
        for (int e = 0; e < NUM_EXPERTS; ++e) acc[e] += xv * wr[e];
    }
#pragma unroll
    for (int off = 16; off > 0; off >>= 1)
#pragma unroll
        for (int e = 0; e < NUM_EXPERTS; ++e)
            acc[e] += __shfl_down(acc[e], off, 32);

    if (lane == 0) {
        float lg[NUM_EXPERTS];
        float mx = -1e30f;
#pragma unroll
        for (int e = 0; e < NUM_EXPERTS; ++e) {
            lg[e] = acc[e] + bg[e];
            mx = fmaxf(mx, lg[e]);
        }
#pragma unroll
        for (int e = 0; e < NUM_EXPERTS; ++e) lg[e] = __expf(lg[e] - mx);
        // top-2 of softmax then renormalize == softmax over top-2 numerators
        int i1 = 0;
#pragma unroll
        for (int e = 1; e < NUM_EXPERTS; ++e) if (lg[e] > lg[i1]) i1 = e;
        int i2 = (i1 == 0) ? 1 : 0;
#pragma unroll
        for (int e = 0; e < NUM_EXPERTS; ++e)
            if (e != i1 && lg[e] > lg[i2]) i2 = e;
        float g1 = lg[i1], g2 = lg[i2];
        float inv = 1.0f / (g1 + g2);
        int p1 = atomicAdd(&counts[i1], 1);
        tok_idx[i1 * CAP + p1] = tok;
        tok_w [i1 * CAP + p1] = g1 * inv;
        int p2 = atomicAdd(&counts[i2], 1);
        tok_idx[i2 * CAP + p2] = tok;
        tok_w [i2 * CAP + p2] = g2 * inv;
    }
}

// -------------------------------------------------- routed expert GEMM ------
// Block: 256 threads = 8 waves (wave32). Tile 128(M) x 128(N), K stage 64
// (2 WMMA K-steps), double-buffered LDS staged with async-to-LDS loads.
// Waves arranged 4(M) x 2(N); each wave owns 2x4 16x16 WMMA subtiles.
#define TM  128
#define TN  128
#define TK  64
#define LDK 72   // padded LDS row (bf16 elems): 144 B -> conflict-free, 16B aligned

__global__ __launch_bounds__(256)
void moe_gemm(const unsigned short* __restrict__ xb,
              const unsigned short* __restrict__ Wt,
              const float* __restrict__ be,
              const int* __restrict__ counts,
              const int* __restrict__ tok_idx,
              const float* __restrict__ tok_w,
              float* __restrict__ out) {
    int e = blockIdx.z;
    int cnt = counts[e];
    int m_base = blockIdx.x * TM;
    if (m_base >= cnt) return;                   // dynamic routing: skip empty tiles
    int n_base = blockIdx.y * TN;

    __shared__ __align__(16) unsigned short As[2][TM][LDK];
    __shared__ __align__(16) unsigned short Bs[2][TN][LDK];
    __shared__ int   s_tok[TM];
    __shared__ float s_w[TM];

    int tid  = threadIdx.x;
    int lane = tid & 31;
    int wid  = tid >> 5;
    int wm   = wid >> 1;                         // 0..3 -> M offset wm*32
    int wn   = wid & 1;                          // 0..1 -> N offset wn*64

    const int*   ti = tok_idx + e * CAP;
    const float* tw = tok_w  + e * CAP;
    if (tid < TM) {
        int m = m_base + tid;
        int valid = (m < cnt);
        s_tok[tid] = valid ? ti[m] : 0;
        s_w [tid] = valid ? tw[m] : 0.0f;
    }
    __syncthreads();

    const unsigned short* Wb = Wt + (size_t)e * HIDDEN * HIDDEN;

    // K-invariant staging addresses: 8 threads per row, 16B (8 bf16) chunks,
    // 4 (row,chunk) slots per thread cover the 128x64 stage.
    const unsigned short* aSrc[4];
    const unsigned short* bSrc[4];
    unsigned ldsOff[4];
#pragma unroll
    for (int i = 0; i < 4; ++i) {
        int idx = i * 256 + tid;
        int row = idx >> 3;
        int col = (idx & 7) * 8;
        aSrc[i] = xb + (size_t)s_tok[row] * HIDDEN + col;
        bSrc[i] = Wb + (size_t)(n_base + row) * HIDDEN + col;
        ldsOff[i] = row * LDK + col;
    }

    auto stage = [&](int buf, int k0) {
        unsigned short* aBase = &As[buf][0][0];
        unsigned short* bBase = &Bs[buf][0][0];
#pragma unroll
        for (int i = 0; i < 4; ++i) {
#if HAVE_ASYNC
            __builtin_amdgcn_global_load_async_to_lds_b128(
                (v4i_vs*)(aSrc[i] + k0),
                (lds_v4i_ptr)(aBase + ldsOff[i]),
                0, 0);
            __builtin_amdgcn_global_load_async_to_lds_b128(
                (v4i_vs*)(bSrc[i] + k0),
                (lds_v4i_ptr)(bBase + ldsOff[i]),
                0, 0);
#else
            *(uint4*)(aBase + ldsOff[i]) = *(const uint4*)(aSrc[i] + k0);
            *(uint4*)(bBase + ldsOff[i]) = *(const uint4*)(bSrc[i] + k0);
#endif
        }
    };

    v8f acc[2][4];
#pragma unroll
    for (int mt = 0; mt < 2; ++mt)
#pragma unroll
        for (int nt = 0; nt < 4; ++nt)
            acc[mt][nt] = (v8f){0.f,0.f,0.f,0.f,0.f,0.f,0.f,0.f};

    int half = lane >> 4;        // ISA 16-bit frag layout: lane<16 -> K {0..7,16..23}
    int l16  = lane & 15;        //                         lane>=16 -> K {8..15,24..31}

    // Prologue: stage 0 into buffer 0.
    stage(0, 0);
    wait_async_zero();
    __syncthreads();

    int buf = 0;
    for (int k0 = 0; k0 < HIDDEN; k0 += TK) {
        int knext = k0 + TK;
        if (knext < HIDDEN) stage(buf ^ 1, knext);   // prefetch next stage

#pragma unroll
        for (int ks = 0; ks < 2; ++ks) {
            int kk = ks * 32 + half * 8;
            v16bf afrag[2], bfrag[4];
#pragma unroll
            for (int mt = 0; mt < 2; ++mt) {
                int row = wm * 32 + mt * 16 + l16;
                union { uint4 u[2]; v16bf v; } ua;
                ua.u[0] = *(const uint4*)&As[buf][row][kk];
                ua.u[1] = *(const uint4*)&As[buf][row][kk + 16];
                afrag[mt] = ua.v;
            }
#pragma unroll
            for (int nt = 0; nt < 4; ++nt) {
                int row = wn * 64 + nt * 16 + l16;
                union { uint4 u[2]; v16bf v; } ub;
                ub.u[0] = *(const uint4*)&Bs[buf][row][kk];
                ub.u[1] = *(const uint4*)&Bs[buf][row][kk + 16];
                bfrag[nt] = ub.v;
            }
#pragma unroll
            for (int mt = 0; mt < 2; ++mt)
#pragma unroll
                for (int nt = 0; nt < 4; ++nt)
                    acc[mt][nt] = __builtin_amdgcn_wmma_f32_16x16x32_bf16(
                        /*neg_a=*/false, afrag[mt],
                        /*neg_b=*/false, bfrag[nt],
                        /*c_mod=*/(short)0, acc[mt][nt],
                        /*reuse_a=*/false, /*reuse_b=*/false);
        }

        wait_async_zero();     // my async writes to buf^1 landed in LDS
        __syncthreads();       // everyone's reads of buf done, writes visible
        buf ^= 1;
    }

    // Epilogue: h = relu(acc + be[n]); out[token] += gate * h (atomic combine).
    float bias[4];
#pragma unroll
    for (int nt = 0; nt < 4; ++nt)
        bias[nt] = be[e * HIDDEN + n_base + wn * 64 + nt * 16 + l16];

    float* outN = out + n_base + wn * 64 + l16;
#pragma unroll
    for (int mt = 0; mt < 2; ++mt) {
#pragma unroll
        for (int r = 0; r < 8; ++r) {
            int mloc = wm * 32 + mt * 16 + half * 8 + r;  // C layout: vgpr r, lane half
            int m = m_base + mloc;
            if (m < cnt) {
                float w = s_w[mloc];
                float* orow = outN + (size_t)s_tok[mloc] * HIDDEN;
#pragma unroll
                for (int nt = 0; nt < 4; ++nt) {
                    float h = acc[mt][nt][r] + bias[nt];
                    h = h > 0.0f ? h : 0.0f;
                    atomicAdd(orow + nt * 16, w * h);
                }
            }
        }
    }
}

// ---------------------------------------------------------------------------
extern "C" void kernel_launch(void* const* d_in, const int* in_sizes, int n_in,
                              void* d_out, int out_size, void* d_ws, size_t ws_size,
                              hipStream_t stream) {
    const float* x  = (const float*)d_in[0];   // [N, H]
    const float* Wg = (const float*)d_in[1];   // [H, E]
    const float* bg = (const float*)d_in[2];   // [E]
    const float* We = (const float*)d_in[3];   // [E, H, H]
    const float* be = (const float*)d_in[4];   // [E, H]
    float* out = (float*)d_out;                // [N, H]

    // Workspace layout
    char* ws = (char*)d_ws;
    int*   counts  = (int*)ws;                                   // 8 (pad 256)
    int*   tok_idx = (int*)(ws + 256);                           // E*CAP ints
    float* tok_w   = (float*)(ws + 256 + (size_t)NUM_EXPERTS * CAP * 4);
    unsigned short* xb =
        (unsigned short*)(ws + 256 + (size_t)NUM_EXPERTS * CAP * 8);
    unsigned short* Wt = xb + (size_t)N_TOKENS * HIDDEN;         // E*H*H bf16

    // 1) zero output + counters
    zero_f32<<<2048, 256, 0, stream>>>(out, (size_t)N_TOKENS * HIDDEN);
    zero_f32<<<1, 32, 0, stream>>>((float*)counts, NUM_EXPERTS);

    // 2) x -> bf16
    convert_x<<<4096, 256, 0, stream>>>(x, xb, (size_t)N_TOKENS * HIDDEN / 4);

    // 3) We -> transposed bf16 Wt[e][n][k]
    dim3 gt(HIDDEN / 32, HIDDEN / 32, NUM_EXPERTS);
    transpose_convert_w<<<gt, dim3(32, 8), 0, stream>>>(We, Wt);

    // 4) gating + routing lists
    gating_kernel<<<(N_TOKENS * 32) / 128, 128, 0, stream>>>(
        x, Wg, bg, counts, tok_idx, tok_w);

    // 5) routed grouped GEMM with bf16 WMMA
    dim3 gg(N_TOKENS / TM, HIDDEN / TN, NUM_EXPERTS);
    moe_gemm<<<gg, 256, 0, stream>>>(xb, Wt, be, counts, tok_idx, tok_w, out);
}
